// DeepGGALayer_29137058136488
// MI455X (gfx1250) — matrix operand order repaired and checked by reading, not compiled, not run on hardware
//
#include <hip/hip_runtime.h>
#include <hip/hip_fp16.h>

#define N_NODES 100000
#define N_EDGES 1600000
#define CH 64
#define CH2 128
#define EPSV 1e-7f
#define BN_EPS 1e-5f

typedef __attribute__((ext_vector_type(16))) _Float16 v16h;
typedef __attribute__((ext_vector_type(8)))  _Float16 v8h;
typedef __attribute__((ext_vector_type(8)))  float    v8f;

// ---------------------------------------------------------------------------
// sign-aware float atomic max (works for mixed signs with -inf init)
__device__ __forceinline__ void atomicMaxF(float* addr, float val) {
    if (val >= 0.0f) atomicMax((int*)addr, __float_as_int(val));
    else             atomicMin((unsigned int*)addr, __float_as_uint(val));
}

// ---------------------------------------------------------------------------
// init per-block scratch: maxb=-inf, denom=0, num=0, stats=0
__global__ void k_init(float* maxb, float* denom, float* num, float* stats) {
    const size_t NC = (size_t)N_NODES * CH;
    size_t stride = (size_t)gridDim.x * blockDim.x;
    for (size_t i = blockIdx.x * (size_t)blockDim.x + threadIdx.x; i < NC; i += stride) {
        maxb[i]  = -INFINITY;
        denom[i] = 0.0f;
        num[i]   = 0.0f;
    }
    size_t gt = blockIdx.x * (size_t)blockDim.x + threadIdx.x;
    if (gt < 1024) stats[gt] = 0.0f;
}

// convert this block's weights to f16
__global__ void k_convW(const float* __restrict__ W1, const float* __restrict__ W2,
                        _Float16* __restrict__ W1h, _Float16* __restrict__ W2h) {
    int i = blockIdx.x * blockDim.x + threadIdx.x;   // 0..16383
    if (i < CH * CH2)      W1h[i] = (_Float16)W1[i];
    else if (i < 2*CH*CH2) W2h[i - CH*CH2] = (_Float16)W2[i - CH*CH2];
}

// ---------------------------------------------------------------------------
// edge pass 1: per-dst per-channel max of logits = (relu(x[src])+eps)*t
__global__ void k_edge_max(const float* __restrict__ x, const int* __restrict__ ei,
                           const float* __restrict__ tptr, float* __restrict__ maxb) {
    long long idx = blockIdx.x * (long long)blockDim.x + threadIdx.x;
    if (idx >= (long long)N_EDGES * 16) return;
    int e  = (int)(idx >> 4);
    int c4 = (int)(idx & 15) << 2;
    int s = ei[e], d = ei[N_EDGES + e];
    float tv = tptr[0];
    float4 xv = *(const float4*)(x + (size_t)s * CH + c4);
    float* mp = maxb + (size_t)d * CH + c4;
    atomicMaxF(mp + 0, (fmaxf(xv.x, 0.f) + EPSV) * tv);
    atomicMaxF(mp + 1, (fmaxf(xv.y, 0.f) + EPSV) * tv);
    atomicMaxF(mp + 2, (fmaxf(xv.z, 0.f) + EPSV) * tv);
    atomicMaxF(mp + 3, (fmaxf(xv.w, 0.f) + EPSV) * tv);
}

// edge pass 2: denom += e, num += msg*e with e = exp(msg*t - max)
__global__ void k_edge_sum(const float* __restrict__ x, const int* __restrict__ ei,
                           const float* __restrict__ tptr, const float* __restrict__ maxb,
                           float* __restrict__ denom, float* __restrict__ num) {
    long long idx = blockIdx.x * (long long)blockDim.x + threadIdx.x;
    if (idx >= (long long)N_EDGES * 16) return;
    int e  = (int)(idx >> 4);
    int c4 = (int)(idx & 15) << 2;
    int s = ei[e], d = ei[N_EDGES + e];
    float tv = tptr[0];
    float4 xv = *(const float4*)(x + (size_t)s * CH + c4);
    float4 mx = *(const float4*)(maxb + (size_t)d * CH + c4);
    float* dp = denom + (size_t)d * CH + c4;
    float* np = num   + (size_t)d * CH + c4;
    float m0 = fmaxf(xv.x,0.f)+EPSV, m1 = fmaxf(xv.y,0.f)+EPSV;
    float m2 = fmaxf(xv.z,0.f)+EPSV, m3 = fmaxf(xv.w,0.f)+EPSV;
    float e0 = __expf(m0*tv - mx.x), e1 = __expf(m1*tv - mx.y);
    float e2 = __expf(m2*tv - mx.z), e3 = __expf(m3*tv - mx.w);
    atomicAdd(dp+0, e0); atomicAdd(dp+1, e1); atomicAdd(dp+2, e2); atomicAdd(dp+3, e3);
    atomicAdd(np+0, m0*e0); atomicAdd(np+1, m1*e1); atomicAdd(np+2, m2*e2); atomicAdd(np+3, m3*e3);
}

// ---------------------------------------------------------------------------
// per-node MessageNorm: out = x + scale*||x|| * agg/max(||agg||,1e-12); f16 out
// one wave32 per node, lane handles channels {lane, lane+32}
__global__ void k_msgnorm(const float* __restrict__ x, const float* __restrict__ denom,
                          const float* __restrict__ num, const float* __restrict__ sptr,
                          _Float16* __restrict__ outh) {
    int wave = threadIdx.x >> 5;
    int lane = threadIdx.x & 31;
    int node = blockIdx.x * 8 + wave;
    if (node >= N_NODES) return;
    size_t base = (size_t)node * CH;
    float d0 = denom[base + lane], d1 = denom[base + lane + 32];
    float n0 = num[base + lane],   n1 = num[base + lane + 32];
    float a0 = d0 > 0.f ? n0 / d0 : 0.f;
    float a1 = d1 > 0.f ? n1 / d1 : 0.f;
    float xv0 = x[base + lane], xv1 = x[base + lane + 32];
    float asq = a0*a0 + a1*a1;
    float xsq = xv0*xv0 + xv1*xv1;
    #pragma unroll
    for (int m = 16; m >= 1; m >>= 1) {
        asq += __shfl_xor(asq, m, 32);
        xsq += __shfl_xor(xsq, m, 32);
    }
    float inv = 1.0f / fmaxf(sqrtf(asq), 1e-12f);
    float f = sptr[0] * sqrtf(xsq) * inv;
    outh[base + lane]      = (_Float16)(xv0 + a0 * f);
    outh[base + lane + 32] = (_Float16)(xv1 + a1 * f);
}

// ---------------------------------------------------------------------------
// GEMM1: [N,64](f16) x [64,128](f16) + b1 -> h0 [N,128] f32.
// One block = 8 waves; wave w computes 16x16 tile at cols [16w,16w+16); 2 WMMAs (K=64)
__global__ void k_gemm1(const _Float16* __restrict__ Ah, const _Float16* __restrict__ Bh,
                        const float* __restrict__ bias, float* __restrict__ H) {
    int tile = blockIdx.x;                 // 6250 row tiles (N = 16*6250 exactly)
    int wave = threadIdx.x >> 5;           // col tile 0..7
    int lane = threadIdx.x & 31;
    int l15 = lane & 15, hi = lane >> 4;
    int row = tile * 16 + l15;
    int col = wave * 16 + l15;

    v8f c; float bv = bias[col];
    #pragma unroll
    for (int v = 0; v < 8; v++) c[v] = bv;

    #pragma unroll
    for (int f = 0; f < 2; f++) {
        // A fragment: lane<16 -> K = f*32 + 0..15; lane>=16 -> K = f*32 + 16..31
        const _Float16* ap = Ah + (size_t)row * CH + f * 32 + hi * 16;
        v8h alo = *(const v8h*)(ap);
        v8h ahi = *(const v8h*)(ap + 8);
        v16h a = __builtin_shufflevector(alo, ahi, 0,1,2,3,4,5,6,7,8,9,10,11,12,13,14,15);
        // B fragment: lane<16 -> K = f*32 + 0..15; lane>=16 -> K = f*32 + 16..31
        int kb = f * 32 + hi * 16;
        v16h b;
        #pragma unroll
        for (int k = 0; k < 16; k++) b[k] = Bh[(size_t)(kb + k) * CH2 + col];
        c = __builtin_amdgcn_wmma_f32_16x16x32_f16(false, a, false, b, (short)0, c, false, false);
    }
    #pragma unroll
    for (int v = 0; v < 8; v++) {
        int m = v + hi * 8;
        H[(size_t)(tile * 16 + m) * CH2 + col] = c[v];
    }
}

// GEMM2: [N,128](f16) x [128,64](f16) + b2 -> y [N,64] f32. 4 waves, 4 WMMAs (K=128)
__global__ void k_gemm2(const _Float16* __restrict__ Ah, const _Float16* __restrict__ Bh,
                        const float* __restrict__ bias, float* __restrict__ Y) {
    int tile = blockIdx.x;
    int wave = threadIdx.x >> 5;           // col tile 0..3
    int lane = threadIdx.x & 31;
    int l15 = lane & 15, hi = lane >> 4;
    int row = tile * 16 + l15;
    int col = wave * 16 + l15;

    v8f c; float bv = bias[col];
    #pragma unroll
    for (int v = 0; v < 8; v++) c[v] = bv;

    #pragma unroll
    for (int f = 0; f < 4; f++) {
        const _Float16* ap = Ah + (size_t)row * CH2 + f * 32 + hi * 16;
        v8h alo = *(const v8h*)(ap);
        v8h ahi = *(const v8h*)(ap + 8);
        v16h a = __builtin_shufflevector(alo, ahi, 0,1,2,3,4,5,6,7,8,9,10,11,12,13,14,15);
        int kb = f * 32 + hi * 16;
        v16h b;
        #pragma unroll
        for (int k = 0; k < 16; k++) b[k] = Bh[(size_t)(kb + k) * CH + col];
        c = __builtin_amdgcn_wmma_f32_16x16x32_f16(false, a, false, b, (short)0, c, false, false);
    }
    #pragma unroll
    for (int v = 0; v < 8; v++) {
        int m = v + hi * 8;
        Y[(size_t)(tile * 16 + m) * CH + col] = c[v];
    }
}

// ---------------------------------------------------------------------------
// column sum / sumsq (blockDim.x == cols)
__global__ void k_colstats(const float* __restrict__ M, int cols,
                           float* __restrict__ sum, float* __restrict__ sq) {
    int col = threadIdx.x;
    float s = 0.f, q = 0.f;
    for (int r = blockIdx.x; r < N_NODES; r += gridDim.x) {
        float v = M[(size_t)r * cols + col];
        s += v; q += v * v;
    }
    atomicAdd(&sum[col], s);
    atomicAdd(&sq[col], q);
}

// BN coefficients: cA = g*rsqrt(var+eps), cB = be - mu*cA
__global__ void k_bncoefs(const float* __restrict__ sum, const float* __restrict__ sq,
                          const float* __restrict__ g, const float* __restrict__ be,
                          float* __restrict__ cA, float* __restrict__ cB, int cols) {
    int c = threadIdx.x;
    if (c >= cols) return;
    float invN = 1.0f / (float)N_NODES;
    float mu = sum[c] * invN;
    float var = sq[c] * invN - mu * mu;
    float a = g[c] * rsqrtf(var + BN_EPS);
    cA[c] = a;
    cB[c] = be[c] - mu * a;
}

// apply BN + relu to h0 -> f16 hh
__global__ void k_applyH(const float* __restrict__ H, const float* __restrict__ cA,
                         const float* __restrict__ cB, _Float16* __restrict__ Hh) {
    size_t idx = blockIdx.x * (size_t)blockDim.x + threadIdx.x;
    if (idx >= (size_t)N_NODES * CH2) return;
    int c = (int)(idx & (CH2 - 1));
    Hh[idx] = (_Float16)fmaxf(H[idx] * cA[c] + cB[c], 0.0f);
}

// apply outer BN (+optional relu+eps) to y -> next x
__global__ void k_applyOut(const float* __restrict__ Y, const float* __restrict__ cA,
                           const float* __restrict__ cB, float* __restrict__ Xn, int do_relu) {
    size_t idx = blockIdx.x * (size_t)blockDim.x + threadIdx.x;
    if (idx >= (size_t)N_NODES * CH) return;
    int c = (int)(idx & (CH - 1));
    float v = Y[idx] * cA[c] + cB[c];
    if (do_relu) v = fmaxf(v, 0.0f) + EPSV;
    Xn[idx] = v;
}

// final: out = relu(x_in + x2) + eps
__global__ void k_final(const float* __restrict__ Xin, const float* __restrict__ X2,
                        float* __restrict__ Out) {
    size_t idx = blockIdx.x * (size_t)blockDim.x + threadIdx.x;
    if (idx >= (size_t)N_NODES * CH) return;
    Out[idx] = fmaxf(Xin[idx] + X2[idx], 0.0f) + EPSV;
}

// ---------------------------------------------------------------------------
extern "C" void kernel_launch(void* const* d_in, const int* in_sizes, int n_in,
                              void* d_out, int out_size, void* d_ws, size_t ws_size,
                              hipStream_t stream) {
    const float* x_in  = (const float*)d_in[0];
    const int*   ei    = (const int*)d_in[1];
    const float* t     = (const float*)d_in[2];
    const float* scale = (const float*)d_in[3];
    const float* W1    = (const float*)d_in[4];
    const float* b1    = (const float*)d_in[5];
    const float* g1    = (const float*)d_in[6];
    const float* be1   = (const float*)d_in[7];
    const float* W2    = (const float*)d_in[8];
    const float* b2    = (const float*)d_in[9];
    const float* bn_g  = (const float*)d_in[10];
    const float* bn_b  = (const float*)d_in[11];
    float* out = (float*)d_out;

    const size_t NC = (size_t)N_NODES * CH;
    float* fws   = (float*)d_ws;
    float* xA    = fws;               // NC  (block0 output)
    float* xB    = xA + NC;           // NC  (block1 output)
    float* maxb  = xB + NC;           // NC
    float* denom = maxb + NC;         // NC
    float* num   = denom + NC;        // NC
    float* h0    = maxb;              // 2*NC  (aliases maxb+denom; dead by GEMM1)
    float* yb    = num;               // NC    (aliases num; dead by GEMM2)
    float* stats = num + NC;          // 1024 floats
    float* sum1 = stats,       *sq1 = stats + 128;
    float* cA1  = stats + 256, *cB1 = stats + 384;
    float* sum2 = stats + 512, *sq2 = stats + 576;
    float* cA2  = stats + 640, *cB2 = stats + 704;
    _Float16* outh = (_Float16*)(stats + 1024);  // NC halves
    _Float16* hh   = outh + NC;                  // 2*NC halves
    _Float16* W1h  = hh + 2 * NC;                // 8192
    _Float16* W2h  = W1h + CH * CH2;             // 8192

    const int TB = 256;
    const int edgeBlocks = (int)(((long long)N_EDGES * 16 + TB - 1) / TB);

    for (int i = 0; i < 2; i++) {
        const float* xcur = (i == 0) ? x_in : xA;
        float* xnext = (i == 0) ? xA : xB;

        k_init<<<8192, TB, 0, stream>>>(maxb, denom, num, stats);
        k_convW<<<(2 * CH * CH2 + TB - 1) / TB, TB, 0, stream>>>(
            W1 + (size_t)i * CH * CH2, W2 + (size_t)i * CH2 * CH, W1h, W2h);

        k_edge_max<<<edgeBlocks, TB, 0, stream>>>(xcur, ei, t + i, maxb);
        k_edge_sum<<<edgeBlocks, TB, 0, stream>>>(xcur, ei, t + i, maxb, denom, num);
        k_msgnorm<<<(N_NODES + 7) / 8, TB, 0, stream>>>(xcur, denom, num, scale + i, outh);

        k_gemm1<<<N_NODES / 16, 256, 0, stream>>>(outh, W1h, b1 + (size_t)i * CH2, h0);
        k_colstats<<<512, CH2, 0, stream>>>(h0, CH2, sum1, sq1);
        k_bncoefs<<<1, CH2, 0, stream>>>(sum1, sq1, g1 + (size_t)i * CH2,
                                         be1 + (size_t)i * CH2, cA1, cB1, CH2);
        k_applyH<<<(int)((NC * 2 + TB - 1) / TB), TB, 0, stream>>>(h0, cA1, cB1, hh);

        k_gemm2<<<N_NODES / 16, 128, 0, stream>>>(hh, W2h, b2 + (size_t)i * CH, yb);
        k_colstats<<<512, CH, 0, stream>>>(yb, CH, sum2, sq2);
        k_bncoefs<<<1, CH, 0, stream>>>(sum2, sq2, bn_g + (size_t)i * CH,
                                        bn_b + (size_t)i * CH, cA2, cB2, CH);
        k_applyOut<<<(int)((NC + TB - 1) / TB), TB, 0, stream>>>(yb, cA2, cB2, xnext,
                                                                 (i == 0) ? 1 : 0);
    }
    k_final<<<(int)((NC + TB - 1) / TB), TB, 0, stream>>>(x_in, xB, out);
}
